// RobustCWA_75505525064180
// MI455X (gfx1250) — matrix operand order repaired and checked by reading, not compiled, not run on hardware
//
#include <hip/hip_runtime.h>
#include <hip/hip_bf16.h>
#include <math.h>

// ---------------------------------------------------------------------------
// Types for CDNA5 WMMA (gfx1250, wave32)
// ---------------------------------------------------------------------------
typedef __bf16 bf16;
typedef __attribute__((ext_vector_type(16))) __bf16 v16bf;
typedef __attribute__((ext_vector_type(8)))  float  v8f;
typedef int v4i_ __attribute__((vector_size(16)));   // matches builtin param

#define D_IN  768
#define D_HID 256
#define D_OUT 64

// Async global->LDS path (CDNA5 GLOBAL_LOAD_ASYNC_TO_LDS_B128, ASYNCcnt)
#define CDNA5_ASYNC_LDS 0
#if defined(__has_builtin)
#if __has_builtin(__builtin_amdgcn_global_load_async_to_lds_b128)
#undef CDNA5_ASYNC_LDS
#define CDNA5_ASYNC_LDS 1
#endif
#endif

__device__ __forceinline__ void async_wait_all() {
#if CDNA5_ASYNC_LDS
#if defined(__has_builtin) && __has_builtin(__builtin_amdgcn_s_wait_asynccnt)
  __builtin_amdgcn_s_wait_asynccnt(0);
#else
  asm volatile("s_wait_asynccnt 0" ::: "memory");
#endif
#endif
}

// D = A(16x32 bf16) * B(32x16 bf16) + C(16x16 f32)
__device__ __forceinline__ v8f wmma_bf(v16bf a, v16bf b, v8f c) {
  return __builtin_amdgcn_wmma_f32_16x16x32_bf16(false, a, false, b, (short)0, c,
                                                 false, false);
}

// B-fragment: B[k][n] = W[n][k] (W row-major, N x ldk, bf16).
// 16-bit B 32x16 layout: lane l -> n = n0+(l&15), kb = k0+(l>>4)*8,
// K = kb..kb+7 in elems 0..7 and kb+16..kb+23 in elems 8..15 (two aligned 16B loads).
__device__ __forceinline__ v16bf load_bfrag(const bf16* W, int ldk, int n0, int k0,
                                            int lane) {
  typedef __attribute__((ext_vector_type(8))) __bf16 v8bf_;
  int n  = n0 + (lane & 15);
  int kb = k0 + ((lane >> 4) << 3);
  const bf16* p = W + (size_t)n * ldk + kb;
  v8bf_ lo = *(const v8bf_*)(p);
  v8bf_ hi = *(const v8bf_*)(p + 16);
  return __builtin_shufflevector(lo, hi, 0, 1, 2, 3, 4, 5, 6, 7, 8, 9, 10, 11, 12, 13, 14, 15);
}

// A-fragment from bf16 row-major buffer (LDS): lane l -> row r0+(l&15),
// K = kb..kb+7 in elems 0..7 and kb+16..kb+23 in elems 8..15, kb = k0+(l>>4)*8.
__device__ __forceinline__ v16bf load_afrag_bf(const bf16* H, int ldk, int r0,
                                               int k0, int lane) {
  const bf16* p = H + (size_t)(r0 + (lane & 15)) * ldk + k0 + ((lane >> 4) << 3);
  v16bf a;
#pragma unroll
  for (int i = 0; i < 8; ++i) { a[i] = p[i]; a[8 + i] = p[16 + i]; }
  return a;
}

// A-fragment from fp32 row-major buffer (LDS or global), converted to bf16.
__device__ __forceinline__ v16bf load_afrag_f32(const float* X, int ldk, int r0,
                                                int k0, int lane) {
  const float* p = X + (size_t)(r0 + (lane & 15)) * ldk + k0 + ((lane >> 4) << 3);
  float4 x0 = *(const float4*)(p);
  float4 x1 = *(const float4*)(p + 4);
  float4 y0 = *(const float4*)(p + 16);
  float4 y1 = *(const float4*)(p + 20);
  v16bf a;
  a[0] = (bf16)x0.x; a[1] = (bf16)x0.y; a[2]  = (bf16)x0.z; a[3]  = (bf16)x0.w;
  a[4] = (bf16)x1.x; a[5] = (bf16)x1.y; a[6]  = (bf16)x1.z; a[7]  = (bf16)x1.w;
  a[8] = (bf16)y0.x; a[9] = (bf16)y0.y; a[10] = (bf16)y0.z; a[11] = (bf16)y0.w;
  a[12]= (bf16)y1.x; a[13]= (bf16)y1.y; a[14] = (bf16)y1.z; a[15] = (bf16)y1.w;
  return a;
}

// ---------------------------------------------------------------------------
// K0: convert weights fp32 -> bf16 workspace copies
// ---------------------------------------------------------------------------
__global__ void k0_prep(const float* W1, const float* W2, const float* Wr,
                        const float* Wo, bf16* w1b, bf16* w2b, bf16* wrb,
                        bf16* wob) {
  const int N1 = D_HID * D_IN, N2 = D_OUT * D_HID, N3 = D_OUT * D_IN,
            N4 = D_OUT * D_OUT;
  const int NT = N1 + N2 + N3 + N4;
  for (int t = blockIdx.x * blockDim.x + threadIdx.x; t < NT;
       t += gridDim.x * blockDim.x) {
    if (t < N1)                w1b[t] = (bf16)W1[t];
    else if (t < N1 + N2)      w2b[t - N1] = (bf16)W2[t - N1];
    else if (t < N1 + N2 + N3) wrb[t - N1 - N2] = (bf16)Wr[t - N1 - N2];
    else                       wob[t - N1 - N2 - N3] = (bf16)Wo[t - N1 - N2 - N3];
  }
}

// ---------------------------------------------------------------------------
// K1: fused  z@W1^T+b1 -> LN -> GELU -> @W2^T + z@Wr^T + biases -> @Wo^T
//     writes h_orth (fp32) and per-workgroup covariance partials.
// 256 threads = 8 waves: wave = (stripe s in {0,1}) x (colgroup c in {0..3}).
// 4 row-blocks of 32 rows per workgroup (128 rows/WG, grid = B/128).
// z tiles staged global->LDS (async on CDNA5), double-buffered: one barrier
// per 32-wide K step.
// ---------------------------------------------------------------------------
__global__ void __launch_bounds__(256)
k1_fused(const float* __restrict__ z, const bf16* __restrict__ W1b,
         const float* __restrict__ b1, const float* __restrict__ ln_g,
         const float* __restrict__ ln_b, const bf16* __restrict__ W2b,
         const float* __restrict__ b2, const bf16* __restrict__ Wrb,
         const float* __restrict__ br, const bf16* __restrict__ Wob,
         float* __restrict__ Xout, float* __restrict__ Spart,
         float* __restrict__ CSpart) {
  __shared__ float za[2][32][32];     // staged z tile (fp32, double buffered)
  __shared__ bf16  hbuf[32][256];     // h after GEMM1 (then LN+GELU, in place)
  __shared__ bf16  hl[32][64];        // h_latent (bf16)
  __shared__ float ho[32][64];        // h_orth (fp32) for covariance
  __shared__ float Sp[4096];          // per-WG covariance partial
  __shared__ float csum[64];
  __shared__ float pesum[32][8], pesq[32][8];
  __shared__ float mean_s[32], rstd_s[32];

  const int tid  = threadIdx.x;
  const int wave = tid >> 5, lane = tid & 31;
  const int s = wave >> 2;            // stripe (16 rows)
  const int c = wave & 3;             // column group
  const int l15 = lane & 15, lh = lane >> 4;
  const int zr = tid >> 3, zq = (tid & 7) << 2;  // staging coords (16B/thread)

  for (int i = tid; i < 4096; i += 256) Sp[i] = 0.f;
  if (tid < 64) csum[tid] = 0.f;
  __syncthreads();

  for (int rb = 0; rb < 4; ++rb) {
    const int rows0 = (blockIdx.x * 4 + rb) * 32;

    // stage one 32x32 fp32 z tile into za[buf] (16B per thread)
    auto stage_z = [&](int buf, int kt) {
      const float* gp = z + (size_t)(rows0 + zr) * D_IN + kt + zq;
#if CDNA5_ASYNC_LDS
      // (global_src, lds_dst, imm_offset, imm_cpol) — same convention as
      // __builtin_amdgcn_global_load_lds; per-lane 16B async copy.
      __builtin_amdgcn_global_load_async_to_lds_b128(
          (v4i_*)gp, (v4i_*)&za[buf][zr][zq], 0, 0);
#else
      *(float4*)&za[buf][zr][zq] = *(const float4*)gp;
#endif
    };

    // L2 prefetch of the next row-block of z (gfx1250 global_prefetch_b8)
    if (rb + 1 < 4)
      __builtin_prefetch(z + (size_t)(rows0 + 32 + zr) * D_IN + zq, 0, 1);

    // ---- stage 1: acc_h = z@W1^T (16x256 per stripe, 4 tiles per wave),
    //               acc_r = z@Wr^T (16x16 tile per wave) -------------------
    v8f acch[4] = {};
    v8f accr    = {};
    stage_z(0, 0);
    async_wait_all();
    __syncthreads();
    for (int ktile = 0; ktile < D_IN / 32; ++ktile) {
      const int buf = ktile & 1;
      if (ktile + 1 < D_IN / 32) stage_z(buf ^ 1, (ktile + 1) * 32);
      const int kt = ktile * 32;
      v16bf a = load_afrag_f32(&za[buf][0][0], 32, s * 16, 0, lane);
#pragma unroll
      for (int t = 0; t < 4; ++t) {
        v16bf bw = load_bfrag(W1b, D_IN, c * 64 + t * 16, kt, lane);
        acch[t] = wmma_bf(a, bw, acch[t]);
      }
      v16bf bwr = load_bfrag(Wrb, D_IN, c * 16, kt, lane);
      accr = wmma_bf(a, bwr, accr);
      async_wait_all();   // my async stores into buf^1 are in LDS
      __syncthreads();    // everyone done reading buf / writing buf^1
    }

    // ---- stage 2: h + b1 -> LDS (bf16) ------------------------------------
#pragma unroll
    for (int t = 0; t < 4; ++t) {
      int n = c * 64 + t * 16 + l15;
      float bb = b1[n];
#pragma unroll
      for (int j = 0; j < 8; ++j)
        hbuf[s * 16 + lh * 8 + j][n] = (bf16)(acch[t][j] + bb);
    }
    __syncthreads();

    // ---- LayerNorm stats (8 threads per row) ------------------------------
    {
      int r = tid >> 3, seg = tid & 7;
      float sm = 0.f, sq = 0.f;
#pragma unroll 4
      for (int n = seg * 32; n < seg * 32 + 32; ++n) {
        float v = (float)hbuf[r][n];
        sm += v; sq += v * v;
      }
      pesum[r][seg] = sm; pesq[r][seg] = sq;
    }
    __syncthreads();
    if (tid < 32) {
      float sm = 0.f, sq = 0.f;
#pragma unroll
      for (int q = 0; q < 8; ++q) { sm += pesum[tid][q]; sq += pesq[tid][q]; }
      float mu = sm * (1.f / 256.f);
      float var = sq * (1.f / 256.f) - mu * mu;
      mean_s[tid] = mu;
      rstd_s[tid] = rsqrtf(var + 1e-5f);
    }
    __syncthreads();

    // ---- normalize + affine + exact GELU (in place) -----------------------
    {
      int r = tid >> 3, nb = (tid & 7) * 32;
      float mu = mean_s[r], rs = rstd_s[r];
#pragma unroll 4
      for (int e = 0; e < 32; ++e) {
        int n = nb + e;
        float x = ((float)hbuf[r][n] - mu) * rs * ln_g[n] + ln_b[n];
        float g = 0.5f * x * (1.f + erff(x * 0.70710678f));
        hbuf[r][n] = (bf16)g;
      }
    }
    __syncthreads();

    // ---- stage 3: h_latent = gelu(h)@W2^T + b2 + (z@Wr^T + br) ------------
    v8f acc2 = accr;
    for (int kt = 0; kt < D_HID; kt += 32) {
      v16bf a  = load_afrag_bf(&hbuf[0][0], 256, s * 16, kt, lane);
      v16bf bw = load_bfrag(W2b, D_HID, c * 16, kt, lane);
      acc2 = wmma_bf(a, bw, acc2);
    }
    {
      int n = c * 16 + l15;
      float bb = b2[n] + br[n];
#pragma unroll
      for (int j = 0; j < 8; ++j)
        hl[s * 16 + lh * 8 + j][n] = (bf16)(acc2[j] + bb);
    }
    __syncthreads();

    // ---- stage 4: h_orth = h_latent @ Wo^T --------------------------------
    v8f acc3 = {};
    for (int kt = 0; kt < D_OUT; kt += 32) {
      v16bf a  = load_afrag_bf(&hl[0][0], 64, s * 16, kt, lane);
      v16bf bw = load_bfrag(Wob, D_OUT, c * 16, kt, lane);
      acc3 = wmma_bf(a, bw, acc3);
    }
    {
      int n = c * 16 + l15;
#pragma unroll
      for (int j = 0; j < 8; ++j) {
        int lr = s * 16 + lh * 8 + j;
        float v = acc3[j];
        Xout[(size_t)(rows0 + lr) * 64 + n] = v;
        ho[lr][n] = v;
      }
    }
    __syncthreads();

    // ---- covariance partial: Sp += ho^T ho, csum += colsum(ho) ------------
#pragma unroll 1
    for (int e = 0; e < 16; ++e) {
      int idx = e * 256 + tid;
      int n = idx >> 6, m = idx & 63;
      float a = 0.f;
#pragma unroll 8
      for (int r = 0; r < 32; ++r) a += ho[r][n] * ho[r][m];
      Sp[idx] += a;
    }
    if (tid < 64) {
      float a = 0.f;
#pragma unroll 8
      for (int r = 0; r < 32; ++r) a += ho[r][tid];
      csum[tid] += a;
    }
    __syncthreads();
  }

  for (int i = tid; i < 4096; i += 256)
    Spart[(size_t)blockIdx.x * 4096 + i] = Sp[i];
  if (tid < 64) CSpart[(size_t)blockIdx.x * 64 + tid] = csum[tid];
}

// ---------------------------------------------------------------------------
// K2: deterministic reduction of per-WG partials
// ---------------------------------------------------------------------------
__global__ void k2_reduce(const float* __restrict__ Spart,
                          const float* __restrict__ CSpart,
                          float* __restrict__ S, float* __restrict__ CS,
                          int nparts) {
  int i = blockIdx.x * blockDim.x + threadIdx.x;
  if (i < 4096) {
    float a = 0.f;
    for (int p = 0; p < nparts; ++p) a += Spart[(size_t)p * 4096 + i];
    S[i] = a;
  } else if (i < 4096 + 64) {
    int j = i - 4096;
    float a = 0.f;
    for (int p = 0; p < nparts; ++p) a += CSpart[(size_t)p * 64 + j];
    CS[j] = a;
  }
}

// ---------------------------------------------------------------------------
// K3 (single WG): sigma, trace-norm, 5 Newton-Schulz iters; fold whitening
// into Av=(Wv@W)/sqrt(norm), Am=(Wm@W)/sqrt(norm), cv=bv-mu@Av^T, cm=bm-mu@Am^T
// ---------------------------------------------------------------------------
__global__ void __launch_bounds__(256)
k3_newton(const float* __restrict__ S, const float* __restrict__ CS,
          const float* __restrict__ Wv, const float* __restrict__ bv,
          const float* __restrict__ Wm, const float* __restrict__ bm,
          float Bn, bf16* __restrict__ Avb, bf16* __restrict__ Amb,
          float* __restrict__ cv, float* __restrict__ cm) {
  __shared__ float Ss[64][64], Wl[64][64], T1[64][64], T2[64][64];
  __shared__ float mu[64];
  __shared__ float normsh;
  const int tid = threadIdx.x;

  if (tid < 64) mu[tid] = CS[tid] / Bn;
  __syncthreads();
  const float denom = 1.f / (Bn - 1.f);
  for (int i = tid; i < 4096; i += 256) {
    int n = i >> 6, m = i & 63;
    Ss[n][m] = (S[i] - Bn * mu[n] * mu[m]) * denom + ((n == m) ? 1e-3f : 0.f);
  }
  __syncthreads();
  if (tid == 0) {
    float tr = 0.f;
    for (int i = 0; i < 64; ++i) tr += Ss[i][i];
    normsh = tr * 1.5f + 1e-6f;
  }
  __syncthreads();
  const float inorm = 1.f / normsh;
  for (int i = tid; i < 4096; i += 256) {
    int n = i >> 6, m = i & 63;
    Ss[n][m] *= inorm;
    Wl[n][m] = (n == m) ? 1.f : 0.f;
  }
  __syncthreads();

  for (int it = 0; it < 5; ++it) {
    for (int i = tid; i < 4096; i += 256) {        // T1 = W @ Ss
      int n = i >> 6, m = i & 63; float a = 0.f;
#pragma unroll 8
      for (int k = 0; k < 64; ++k) a += Wl[n][k] * Ss[k][m];
      T1[n][m] = a;
    }
    __syncthreads();
    for (int i = tid; i < 4096; i += 256) {        // T2 = T1 @ W^T  (= P)
      int n = i >> 6, m = i & 63; float a = 0.f;
#pragma unroll 8
      for (int k = 0; k < 64; ++k) a += T1[n][k] * Wl[m][k];
      T2[n][m] = a;
    }
    __syncthreads();
    for (int i = tid; i < 4096; i += 256) {        // T1 = P @ W
      int n = i >> 6, m = i & 63; float a = 0.f;
#pragma unroll 8
      for (int k = 0; k < 64; ++k) a += T2[n][k] * Wl[k][m];
      T1[n][m] = a;
    }
    __syncthreads();
    for (int i = tid; i < 4096; i += 256) {        // W = 1.5 W - 0.5 P W
      int n = i >> 6, m = i & 63;
      Wl[n][m] = 1.5f * Wl[n][m] - 0.5f * T1[n][m];
    }
    __syncthreads();
  }

  const float rs = sqrtf(inorm);                   // 1/sqrt(norm)
  for (int i = tid; i < 4096; i += 256) {
    int n = i >> 6, k = i & 63;
    float av = 0.f, am = 0.f;
#pragma unroll 8
    for (int j = 0; j < 64; ++j) {
      av += Wv[n * 64 + j] * Wl[j][k];
      am += Wm[n * 64 + j] * Wl[j][k];
    }
    av *= rs; am *= rs;
    T1[n][k] = av; T2[n][k] = am;
    Avb[n * 64 + k] = (bf16)av;
    Amb[n * 64 + k] = (bf16)am;
  }
  __syncthreads();
  if (tid < 64) {
    float a = bv[tid], b = bm[tid];
#pragma unroll 8
    for (int k = 0; k < 64; ++k) { a -= mu[k] * T1[tid][k]; b -= mu[k] * T2[tid][k]; }
    cv[tid] = a; cm[tid] = b;
  }
}

// ---------------------------------------------------------------------------
// K4: out = (X@Av^T + cv) * sigmoid(X@Am^T + cm) * scale
// one 16-row stripe per wave, 8 waves/WG
// ---------------------------------------------------------------------------
__global__ void __launch_bounds__(256)
k4_out(const float* __restrict__ X, const bf16* __restrict__ Avb,
       const bf16* __restrict__ Amb, const float* __restrict__ cv,
       const float* __restrict__ cm, const float* __restrict__ scale,
       float* __restrict__ out) {
  const int wave = threadIdx.x >> 5, lane = threadIdx.x & 31;
  const int l15 = lane & 15, lh = lane >> 4;
  const int r0 = blockIdx.x * 128 + wave * 16;

  v8f av[4] = {};
  v8f am[4] = {};
  for (int kt = 0; kt < 64; kt += 32) {
    v16bf a = load_afrag_f32(X, 64, r0, kt, lane);
#pragma unroll
    for (int t = 0; t < 4; ++t) {
      av[t] = wmma_bf(a, load_bfrag(Avb, 64, t * 16, kt, lane), av[t]);
      am[t] = wmma_bf(a, load_bfrag(Amb, 64, t * 16, kt, lane), am[t]);
    }
  }
  const float scl = scale[0];
#pragma unroll
  for (int t = 0; t < 4; ++t) {
    int n = t * 16 + l15;
    float cvn = cv[n], cmn = cm[n];
#pragma unroll
    for (int j = 0; j < 8; ++j) {
      int row = r0 + lh * 8 + j;
      float v  = av[t][j] + cvn;
      float ml = am[t][j] + cmn;
      float sg = 1.f / (1.f + expf(-ml));
      out[(size_t)row * 64 + n] = v * sg * scl;
    }
  }
}

// ---------------------------------------------------------------------------
// Workspace layout (bytes, all offsets 256B-aligned); total ~51.6 MB
// ---------------------------------------------------------------------------
static constexpr size_t OFF_W1B   = 0;         // 256*768 bf16   = 393216
static constexpr size_t OFF_W2B   = 393216;    // 64*256 bf16    = 32768
static constexpr size_t OFF_WRB   = 425984;    // 64*768 bf16    = 98304
static constexpr size_t OFF_WOB   = 524288;    // 64*64 bf16     = 8192
static constexpr size_t OFF_AVB   = 532480;    // 64*64 bf16
static constexpr size_t OFF_AMB   = 540672;    // 64*64 bf16
static constexpr size_t OFF_CV    = 548864;    // 64 f32
static constexpr size_t OFF_CM    = 549120;    // 64 f32
static constexpr size_t OFF_S     = 549376;    // 4096 f32
static constexpr size_t OFF_CS    = 565760;    // 64 f32
static constexpr size_t OFF_X     = 1048576;   // B*64 f32       = 33554432
static constexpr size_t OFF_SPART = 34603008;  // 1024*4096 f32  = 16777216
static constexpr size_t OFF_CSP   = 51380224;  // 1024*64 f32    = 262144

extern "C" void kernel_launch(void* const* d_in, const int* in_sizes, int n_in,
                              void* d_out, int out_size, void* d_ws,
                              size_t ws_size, hipStream_t stream) {
  const float* z     = (const float*)d_in[0];
  const float* W1    = (const float*)d_in[1];
  const float* b1    = (const float*)d_in[2];
  const float* ln_g  = (const float*)d_in[3];
  const float* ln_b  = (const float*)d_in[4];
  const float* W2    = (const float*)d_in[5];
  const float* b2    = (const float*)d_in[6];
  const float* Wr    = (const float*)d_in[7];
  const float* br    = (const float*)d_in[8];
  const float* Wo    = (const float*)d_in[9];
  const float* Wv    = (const float*)d_in[10];
  const float* bv    = (const float*)d_in[11];
  const float* Wm    = (const float*)d_in[12];
  const float* bm    = (const float*)d_in[13];
  const float* scale = (const float*)d_in[14];

  const int B   = in_sizes[0] / D_IN;  // 131072
  const int nwg = B / 128;             // 1024 workgroups for K1/K4

  char* ws = (char*)d_ws;
  bf16*  W1b   = (bf16*)(ws + OFF_W1B);
  bf16*  W2b   = (bf16*)(ws + OFF_W2B);
  bf16*  Wrb   = (bf16*)(ws + OFF_WRB);
  bf16*  Wob   = (bf16*)(ws + OFF_WOB);
  bf16*  Avb   = (bf16*)(ws + OFF_AVB);
  bf16*  Amb   = (bf16*)(ws + OFF_AMB);
  float* cvp   = (float*)(ws + OFF_CV);
  float* cmp   = (float*)(ws + OFF_CM);
  float* Sfull = (float*)(ws + OFF_S);
  float* CS    = (float*)(ws + OFF_CS);
  float* X     = (float*)(ws + OFF_X);
  float* Spart = (float*)(ws + OFF_SPART);
  float* CSp   = (float*)(ws + OFF_CSP);

  k0_prep<<<256, 256, 0, stream>>>(W1, W2, Wr, Wo, W1b, W2b, Wrb, Wob);
  k1_fused<<<nwg, 256, 0, stream>>>(z, W1b, b1, ln_g, ln_b, W2b, b2, Wrb, br,
                                    Wob, X, Spart, CSp);
  k2_reduce<<<17, 256, 0, stream>>>(Spart, CSp, Sfull, CS, nwg);
  k3_newton<<<1, 256, 0, stream>>>(Sfull, CS, Wv, bv, Wm, bm, (float)B, Avb,
                                   Amb, cvp, cmp);
  k4_out<<<nwg, 256, 0, stream>>>(X, Avb, Amb, cvp, cmp, scale, (float*)d_out);
}
